// SelfAttention_59236188946602
// MI455X (gfx1250) — matrix operand order repaired and checked by reading, not compiled, not run on hardware
//
#include <hip/hip_runtime.h>
#include <hip/hip_bf16.h>

typedef __attribute__((ext_vector_type(16))) _Float16 v16h;
typedef __attribute__((ext_vector_type(8)))  _Float16 v8h;
typedef __attribute__((ext_vector_type(8)))  float    v8f;

#define HEADS  16
#define HD     64
#define SEQ    2048
#define DIM    1024
#define NBATCH 2
#define MROWS  (NBATCH * SEQ)   // 4096

// ---------------- WMMA fragment loaders (wave32, v_wmma_f32_16x16x32_f16) ----
// A fragment: 16(M) x 32(K), A row-major, leading dim lda (elements).
// Layout used here (contiguous k order): lane L: m=L%16, kh=L/16; elems 0..15 -> K=16*kh..16*kh+15
__device__ __forceinline__ v16h load_a_frag(const _Float16* A, int lda) {
  const int lane = threadIdx.x & 31;
  const int m  = lane & 15;
  const int kh = lane >> 4;
  const _Float16* p = A + m * lda + 16 * kh;
  v8h lo = *(const v8h*)(p);        // 16B load
  v8h hi = *(const v8h*)(p + 8);    // 16B load
  v16h r;
#pragma unroll
  for (int i = 0; i < 8; ++i) { r[i] = lo[i]; r[8 + i] = hi[i]; }
  return r;
}

// B fragment for C = A @ Bt^T where Bt is stored [N x K] row-major.
// B layout used here (same contiguous k order as A): lane L: n=L%16, kh=L/16 holds K=16*kh..16*kh+15 -> one 32B load.
__device__ __forceinline__ v16h load_bt_frag(const _Float16* Bt, int ldb) {
  const int lane = threadIdx.x & 31;
  const int n  = lane & 15;
  const int kh = lane >> 4;
  return *(const v16h*)(Bt + n * ldb + 16 * kh);
}

__device__ __forceinline__ v8f wmma_f16(v16h a, v16h b, v8f c) {
  // 8 args: (neg_a, A, neg_b, B, c_mod, C, reuse_a, reuse_b)
  return __builtin_amdgcn_wmma_f32_16x16x32_f16(false, a, false, b, (short)0, c,
                                                false, false);
}

// Software-pipelined 16x64 GEMM tile: C += A[16 x KD] @ Bt[64 x KD]^T.
// Explicit 2x-unrolled PING-PONG double buffer; accumulators are four named
// v8f references (not an array) so the allocator can pin each to a stable
// even-aligned VGPR octet across the loop back-edge (avoids phi copies).
template<int KD>
__device__ __forceinline__ void gemm16x64(const _Float16* A, int lda,
                                          const _Float16* Bt, int ldb,
                                          v8f& acc0, v8f& acc1,
                                          v8f& acc2, v8f& acc3) {
  static_assert(KD % 64 == 0, "KD must be a multiple of 64");
  v16h a0  = load_a_frag(A, lda);
  v16h b00 = load_bt_frag(Bt + 0 * 16 * ldb, ldb);
  v16h b01 = load_bt_frag(Bt + 1 * 16 * ldb, ldb);
  v16h b02 = load_bt_frag(Bt + 2 * 16 * ldb, ldb);
  v16h b03 = load_bt_frag(Bt + 3 * 16 * ldb, ldb);
  v16h a1  = load_a_frag(A + 32, lda);
  v16h b10 = load_bt_frag(Bt + 0 * 16 * ldb + 32, ldb);
  v16h b11 = load_bt_frag(Bt + 1 * 16 * ldb + 32, ldb);
  v16h b12 = load_bt_frag(Bt + 2 * 16 * ldb + 32, ldb);
  v16h b13 = load_bt_frag(Bt + 3 * 16 * ldb + 32, ldb);
#pragma unroll 1
  for (int k0 = 0; k0 < KD - 64; k0 += 64) {
    // consume buffer 0 (K = k0), refill it for K = k0+64
    acc0 = wmma_f16(a0, b00, acc0);
    acc1 = wmma_f16(a0, b01, acc1);
    acc2 = wmma_f16(a0, b02, acc2);
    acc3 = wmma_f16(a0, b03, acc3);
    a0  = load_a_frag(A + k0 + 64, lda);
    b00 = load_bt_frag(Bt + 0 * 16 * ldb + k0 + 64, ldb);
    b01 = load_bt_frag(Bt + 1 * 16 * ldb + k0 + 64, ldb);
    b02 = load_bt_frag(Bt + 2 * 16 * ldb + k0 + 64, ldb);
    b03 = load_bt_frag(Bt + 3 * 16 * ldb + k0 + 64, ldb);
    // consume buffer 1 (K = k0+32), refill it for K = k0+96
    acc0 = wmma_f16(a1, b10, acc0);
    acc1 = wmma_f16(a1, b11, acc1);
    acc2 = wmma_f16(a1, b12, acc2);
    acc3 = wmma_f16(a1, b13, acc3);
    a1  = load_a_frag(A + k0 + 96, lda);
    b10 = load_bt_frag(Bt + 0 * 16 * ldb + k0 + 96, ldb);
    b11 = load_bt_frag(Bt + 1 * 16 * ldb + k0 + 96, ldb);
    b12 = load_bt_frag(Bt + 2 * 16 * ldb + k0 + 96, ldb);
    b13 = load_bt_frag(Bt + 3 * 16 * ldb + k0 + 96, ldb);
  }
  // epilogue: consume both buffers (K = KD-64, KD-32)
  acc0 = wmma_f16(a0, b00, acc0);
  acc1 = wmma_f16(a0, b01, acc1);
  acc2 = wmma_f16(a0, b02, acc2);
  acc3 = wmma_f16(a0, b03, acc3);
  acc0 = wmma_f16(a1, b10, acc0);
  acc1 = wmma_f16(a1, b11, acc1);
  acc2 = wmma_f16(a1, b12, acc2);
  acc3 = wmma_f16(a1, b13, acc3);
}

// ---------------- fp32 -> fp16 converter -------------------------------------
__global__ void __launch_bounds__(256) cvt_f32_to_f16(const float* __restrict__ src,
                                                      _Float16* __restrict__ dst, int n) {
  for (int i = blockIdx.x * blockDim.x + threadIdx.x; i < n;
       i += gridDim.x * blockDim.x)
    dst[i] = (_Float16)src[i];
}

// ---------------- QKV projection: qkv = x @ Wqkv^T + bqkv --------------------
// One wave -> 16x64 output tile. Scatter q,k into [b,h,s,64]; V is stored
// TRANSPOSED [b,h,64,s] so the attention P@V B-fragments are K-contiguous.
__global__ void __launch_bounds__(256) qkv_gemm(const _Float16* __restrict__ x16,
                                                const _Float16* __restrict__ w16,
                                                const float* __restrict__ bqkv,
                                                _Float16* __restrict__ q16,
                                                _Float16* __restrict__ k16,
                                                _Float16* __restrict__ v16T) {
  const int wave   = (int)((blockIdx.x * blockDim.x + threadIdx.x) >> 5);
  const int nTiles = (3 * DIM) / 64;   // 48
  const int mT = wave / nTiles;
  const int nT = wave % nTiles;
  if (mT >= MROWS / 16) return;        // wave-uniform; EXEC stays all-ones

  v8f acc0, acc1, acc2, acc3;
#pragma unroll
  for (int i = 0; i < 8; ++i) { acc0[i] = 0.f; acc1[i] = 0.f; acc2[i] = 0.f; acc3[i] = 0.f; }

  gemm16x64<DIM>(x16 + mT * 16 * DIM, DIM, w16 + nT * 64 * DIM, DIM,
                 acc0, acc1, acc2, acc3);

  const int lane = threadIdx.x & 31;
  const int n  = lane & 15;
  const int rh = lane >> 4;
  v8f accs[4] = {acc0, acc1, acc2, acc3};
#pragma unroll
  for (int t = 0; t < 4; ++t) {
    const int   c    = nT * 64 + t * 16 + n;   // global column 0..3071
    const float bias = bqkv[c];
    const int which  = c >> 10;                // 0=q 1=k 2=v (wave-uniform)
    const int dp     = c & (DIM - 1);
    const int head   = dp >> 6;
    const int hh     = dp & (HD - 1);
    if (which < 2) {
      _Float16* dst = (which == 0) ? q16 : k16;
#pragma unroll
      for (int r = 0; r < 8; ++r) {
        const int mg = mT * 16 + r + 8 * rh;   // token row 0..4095
        const int b  = mg >> 11;
        const int tq = mg & (SEQ - 1);
        dst[(((b * HEADS + head) * SEQ) + tq) * HD + hh] = (_Float16)(accs[t][r] + bias);
      }
    } else {
#pragma unroll
      for (int r = 0; r < 8; ++r) {
        const int mg = mT * 16 + r + 8 * rh;
        const int b  = mg >> 11;
        const int tq = mg & (SEQ - 1);
        v16T[(((b * HEADS + head) * HD) + hh) * SEQ + tq] = (_Float16)(accs[t][r] + bias);
      }
    }
  }
}

// ---------------- Flash attention: one wave per (b, h, 16-query tile) --------
__global__ void __launch_bounds__(256) attn_flash(const _Float16* __restrict__ q16,
                                                  const _Float16* __restrict__ k16,
                                                  const _Float16* __restrict__ v16T,
                                                  const int* __restrict__ maskp,
                                                  _Float16* __restrict__ attn16) {
  __shared__ _Float16 plds[8][16 * 32];   // per-wave P transpose bounce (8 KB)
  const int wib  = threadIdx.x >> 5;
  const int wave = blockIdx.x * 8 + wib;
  const int qt = wave & (SEQ / 16 - 1);   // 0..127
  const int bh = wave >> 7;               // 0..31  (b*HEADS + h)
  if (bh >= NBATCH * HEADS) return;

  const _Float16* Q   = q16  + (size_t)bh * SEQ * HD;
  const _Float16* Kp  = k16  + (size_t)bh * SEQ * HD;
  const _Float16* VpT = v16T + (size_t)bh * HD * SEQ;   // [64][2048]
  const int q0 = qt * 16;
  const int lane = threadIdx.x & 31;
  const int n  = lane & 15;
  const int rh = lane >> 4;
  const int causal = (maskp[0] != 0);
  const float scale = 0.125f;             // 1/sqrt(64)

  // Q fragments reused across the whole key sweep (hd=64 -> two K=32 slices)
  v16h qa0 = load_a_frag(Q + q0 * HD + 0, HD);
  v16h qa1 = load_a_frag(Q + q0 * HD + 32, HD);

  v8f o0, o1, o2, o3;
  float mrow[8], lrow[8];
#pragma unroll
  for (int r = 0; r < 8; ++r) {
    o0[r] = o1[r] = o2[r] = o3[r] = 0.f;
    mrow[r] = -__builtin_inff();
    lrow[r] = 0.f;
  }

  _Float16* P = &plds[wib][0];
  const int kmax = causal ? (q0 + 16) : SEQ;
  const int nChunks = (kmax + 31) >> 5;

  for (int cchunk = 0; cchunk < nChunks; ++cchunk) {
    const int k0 = cchunk * 32;

    // Issue ALL chunk loads up front: 4 K-frags + 4 V-frags (all contiguous b128).
    // S-WMMAs wait only on the K loads; softmax VALU overlaps the V loads.
    v16h kb00 = load_bt_frag(Kp + (k0)      * HD + 0,  HD);
    v16h kb01 = load_bt_frag(Kp + (k0)      * HD + 32, HD);
    v16h kb10 = load_bt_frag(Kp + (k0 + 16) * HD + 0,  HD);
    v16h kb11 = load_bt_frag(Kp + (k0 + 16) * HD + 32, HD);
    v16h vb0  = load_bt_frag(VpT + 0 * 16 * SEQ + k0, SEQ);
    v16h vb1  = load_bt_frag(VpT + 1 * 16 * SEQ + k0, SEQ);
    v16h vb2  = load_bt_frag(VpT + 2 * 16 * SEQ + k0, SEQ);
    v16h vb3  = load_bt_frag(VpT + 3 * 16 * SEQ + k0, SEQ);

    // S = Q @ K^T for 32 keys: two 16x16 tiles
    v8f s0, s1;
#pragma unroll
    for (int i = 0; i < 8; ++i) { s0[i] = 0.f; s1[i] = 0.f; }
    s0 = wmma_f16(qa0, kb00, s0);
    s0 = wmma_f16(qa1, kb01, s0);
    s1 = wmma_f16(qa0, kb10, s1);
    s1 = wmma_f16(qa1, kb11, s1);

    // online softmax over the 32-key chunk (row = across 16 lanes of a half-wave)
    float p0[8], p1[8], alpha[8];
#pragma unroll
    for (int r = 0; r < 8; ++r) {
      const int mg = q0 + r + 8 * rh;
      float a0 = (!causal || (k0 + n      <= mg)) ? s0[r] * scale : -__builtin_inff();
      float a1 = (!causal || (k0 + 16 + n <= mg)) ? s1[r] * scale : -__builtin_inff();
      float mx = fmaxf(a0, a1);
#pragma unroll
      for (int off = 1; off < 16; off <<= 1)
        mx = fmaxf(mx, __shfl_xor(mx, off, 16));
      const float mnew = fmaxf(mrow[r], mx);
      alpha[r] = __expf(mrow[r] - mnew);
      mrow[r]  = mnew;
      a0 = __expf(a0 - mnew);
      a1 = __expf(a1 - mnew);
      float sm = a0 + a1;
#pragma unroll
      for (int off = 1; off < 16; off <<= 1)
        sm += __shfl_xor(sm, off, 16);
      lrow[r] = lrow[r] * alpha[r] + sm;
      p0[r] = a0;
      p1[r] = a1;
    }

    // rescale running O
#pragma unroll
    for (int r = 0; r < 8; ++r) {
      o0[r] *= alpha[r]; o1[r] *= alpha[r]; o2[r] *= alpha[r]; o3[r] *= alpha[r];
    }

    // P: C-layout -> A-layout transpose through per-wave LDS (DS ops in-order/wave)
#pragma unroll
    for (int r = 0; r < 8; ++r) {
      const int row = r + 8 * rh;
      P[row * 32 + n]      = (_Float16)p0[r];
      P[row * 32 + 16 + n] = (_Float16)p1[r];
    }
    __builtin_amdgcn_wave_barrier();
    v16h pa = load_a_frag(P, 32);
    __builtin_amdgcn_wave_barrier();

    // O += P @ V using the already-resident transposed-V fragments
    o0 = wmma_f16(pa, vb0, o0);
    o1 = wmma_f16(pa, vb1, o1);
    o2 = wmma_f16(pa, vb2, o2);
    o3 = wmma_f16(pa, vb3, o3);
  }

  // normalize and write attn output as f16 [b, s, h*64+vd] (A-matrix for proj GEMM)
  const int b = bh >> 4;
  const int h = bh & 15;
#pragma unroll
  for (int r = 0; r < 8; ++r) {
    const float inv = 1.0f / lrow[r];
    const int tq = q0 + r + 8 * rh;
    const size_t base = ((size_t)(b * SEQ + tq)) * DIM + h * HD;
    attn16[base + 0  + n] = (_Float16)(o0[r] * inv);
    attn16[base + 16 + n] = (_Float16)(o1[r] * inv);
    attn16[base + 32 + n] = (_Float16)(o2[r] * inv);
    attn16[base + 48 + n] = (_Float16)(o3[r] * inv);
  }
}

// ---------------- Output projection: out = attn @ Wd^T + bd (fp32 out) -------
__global__ void __launch_bounds__(256) out_gemm(const _Float16* __restrict__ a16,
                                                const _Float16* __restrict__ wd16,
                                                const float* __restrict__ bd,
                                                float* __restrict__ out) {
  const int wave   = (int)((blockIdx.x * blockDim.x + threadIdx.x) >> 5);
  const int nTiles = DIM / 64;   // 16
  const int mT = wave / nTiles;
  const int nT = wave % nTiles;
  if (mT >= MROWS / 16) return;

  v8f acc0, acc1, acc2, acc3;
#pragma unroll
  for (int i = 0; i < 8; ++i) { acc0[i] = 0.f; acc1[i] = 0.f; acc2[i] = 0.f; acc3[i] = 0.f; }

  gemm16x64<DIM>(a16 + mT * 16 * DIM, DIM, wd16 + nT * 64 * DIM, DIM,
                 acc0, acc1, acc2, acc3);

  const int lane = threadIdx.x & 31;
  const int n  = lane & 15;
  const int rh = lane >> 4;
  v8f accs[4] = {acc0, acc1, acc2, acc3};
#pragma unroll
  for (int t = 0; t < 4; ++t) {
    const int   c    = nT * 64 + t * 16 + n;
    const float bias = bd[c];
#pragma unroll
    for (int r = 0; r < 8; ++r) {
      const int mg = mT * 16 + r + 8 * rh;
      out[(size_t)mg * DIM + c] = accs[t][r] + bias;
    }
  }
}

// ---------------- host launcher ----------------------------------------------
extern "C" void kernel_launch(void* const* d_in, const int* in_sizes, int n_in,
                              void* d_out, int out_size, void* d_ws, size_t ws_size,
                              hipStream_t stream) {
  const float* x    = (const float*)d_in[0];   // [2,2048,1024]
  const float* Wqkv = (const float*)d_in[1];   // [3072,1024]
  const float* bqkv = (const float*)d_in[2];   // [3072]
  const float* Wd   = (const float*)d_in[3];   // [1024,1024]
  const float* bd   = (const float*)d_in[4];   // [1024]
  const int*   mask = (const int*)d_in[5];     // scalar 1
  float* out = (float*)d_out;

  char* ws = (char*)d_ws;
  size_t off = 0;
  _Float16* x16    = (_Float16*)(ws + off); off += (size_t)MROWS * DIM * 2;      // 8 MB
  _Float16* wqkv16 = (_Float16*)(ws + off); off += (size_t)3 * DIM * DIM * 2;    // 6 MB
  _Float16* wd16   = (_Float16*)(ws + off); off += (size_t)DIM * DIM * 2;        // 2 MB
  _Float16* q16    = (_Float16*)(ws + off); off += (size_t)MROWS * DIM * 2;      // 8 MB
  _Float16* k16    = (_Float16*)(ws + off); off += (size_t)MROWS * DIM * 2;      // 8 MB
  _Float16* v16T   = (_Float16*)(ws + off); off += (size_t)MROWS * DIM * 2;      // 8 MB
  _Float16* attn16 = x16;  // x16 is dead after qkv_gemm; alias to save workspace

  cvt_f32_to_f16<<<2048, 256, 0, stream>>>(x, x16, MROWS * DIM);
  cvt_f32_to_f16<<<2048, 256, 0, stream>>>(Wqkv, wqkv16, 3 * DIM * DIM);
  cvt_f32_to_f16<<<1024, 256, 0, stream>>>(Wd, wd16, DIM * DIM);

  // 12288 waves (256 mTiles x 48 nTiles), 8 waves/block
  qkv_gemm<<<1536, 256, 0, stream>>>(x16, wqkv16, bqkv, q16, k16, v16T);

  // 4096 waves (2 batch x 16 heads x 128 query tiles), 8 waves/block
  attn_flash<<<512, 256, 0, stream>>>(q16, k16, v16T, mask, attn16);

  // 4096 waves (256 mTiles x 16 nTiles), 8 waves/block
  out_gemm<<<512, 256, 0, stream>>>(attn16, wd16, bd, out);
}
